// PointPillars_9964324127114
// MI455X (gfx1250) — matrix operand order repaired and checked by reading, not compiled, not run on hardware
//
#include <hip/hip_runtime.h>
#include <hip/hip_bf16.h>
#include <math.h>

// ---------------------------------------------------------------------------
// PointPillars-style net on MI455X (gfx1250, wave32, WMMA).
// conv1 (131 GFLOP) and conv2 run as implicit-GEMM on v_wmma_f32_16x16x32_bf16.
// Canvas / pooled activations are NHWC bf16 so A-fragments are contiguous
// 16B loads; weights are pre-swizzled into the B-fragment lane layout.
// Training-mode BatchNorm cancels the conv/PFN/FC1 biases (mean subtraction),
// so the conv/PFN adds are omitted, FC1's kept (mathematically identical).
// ---------------------------------------------------------------------------

typedef __bf16 bf16_t;
typedef bf16_t v16bf __attribute__((ext_vector_type(16)));
typedef bf16_t v8bf  __attribute__((ext_vector_type(8)));
typedef float  v8f   __attribute__((ext_vector_type(8)));

static constexpr int BSZ  = 4;
static constexpr int CIN0 = 8;
static constexpr int P_   = 12000;
static constexpr int N_   = 32;
static constexpr int HH   = 282;       // canvas H = W
static constexpr float RANGE_ = 22.0f;
static constexpr float PILLAR_ = 0.16f;
static constexpr float EPS_ = 1e-5f;

static constexpr int OX1 = 278, PO1 = 139;   // conv1 out, pooled1
static constexpr int OX2 = 136, PO2 = 68;    // conv2 out, pooled2
static constexpr long PFN_CNT = (long)BSZ * P_ * N_;        // 1,536,000
static constexpr long C1_CNT  = (long)BSZ * OX1 * OX1;      // per-channel BN count
static constexpr long C2_CNT  = (long)BSZ * OX2 * OX2;
static constexpr int FC_K = 16 * 68 * 68;    // 73984

// ---------------------------------------------------------------- utilities
__global__ void zero_f32(float* p, long n) {
  long i = (long)blockIdx.x * blockDim.x + threadIdx.x;
  if (i < n) p[i] = 0.0f;
}
__global__ void zero_u128(uint4* p, long n) {
  long i = (long)blockIdx.x * blockDim.x + threadIdx.x;
  if (i < n) p[i] = make_uint4(0u, 0u, 0u, 0u);
}

// ------------------------------------------------------------------- PFN
// pass 1: per-channel sum / sumsq of y = W x  (bias cancels in BN)
__global__ __launch_bounds__(256)
void pfn_stats(const float* __restrict__ x, const float* __restrict__ w,
               float* __restrict__ sum, float* __restrict__ sumsq) {
  const int o = blockIdx.y;
  float wl[8];
#pragma unroll
  for (int c = 0; c < 8; ++c) wl[c] = w[o * 8 + c];
  const long PN = (long)P_ * N_;
  long i = (long)blockIdx.x * blockDim.x + threadIdx.x;
  const long stride = (long)gridDim.x * blockDim.x;
  float s = 0.f, s2 = 0.f;
  for (; i < PFN_CNT; i += stride) {
    long b = i / PN; long r = i - b * PN;
    const float* xb = x + b * 8 * PN + r;
    float y = 0.f;
#pragma unroll
    for (int c = 0; c < 8; ++c) y += xb[c * PN] * wl[c];
    s += y; s2 += y * y;
  }
  __shared__ float rs[256], rq[256];
  rs[threadIdx.x] = s; rq[threadIdx.x] = s2;
  __syncthreads();
  for (int st = 128; st > 0; st >>= 1) {
    if (threadIdx.x < st) { rs[threadIdx.x] += rs[threadIdx.x + st];
                            rq[threadIdx.x] += rq[threadIdx.x + st]; }
    __syncthreads();
  }
  if (threadIdx.x == 0) { atomicAdd(&sum[o], rs[0]); atomicAdd(&sumsq[o], rq[0]); }
}

// pass 2: normalize, relu, max over N points -> out[(b*P+p)*64 + o]
__global__ __launch_bounds__(256)
void pfn_apply(const float* __restrict__ x, const float* __restrict__ w,
               const float* __restrict__ g, const float* __restrict__ be,
               const float* __restrict__ sum, const float* __restrict__ sumsq,
               float* __restrict__ out) {
  __shared__ float xs[4][8][32];
  const int ppb = blockIdx.x * 4;
  for (int i = threadIdx.x; i < 4 * 8 * 32; i += 256) {
    int q = i >> 8; int rem = i & 255; int c = rem >> 5; int n = rem & 31;
    int pp = ppb + q; int b = pp / P_; int p = pp - b * P_;
    xs[q][c][n] = x[((size_t)(b * 8 + c) * P_ + p) * 32 + n];
  }
  __syncthreads();
  const int q = threadIdx.x >> 6;
  const int o = threadIdx.x & 63;
  float wl[8];
#pragma unroll
  for (int c = 0; c < 8; ++c) wl[c] = w[o * 8 + c];
  const float invc = 1.0f / (float)PFN_CNT;
  const float mean = sum[o] * invc;
  const float var  = sumsq[o] * invc - mean * mean;
  const float scale = g[o] * rsqrtf(var + EPS_);
  const float shift = be[o] - mean * scale;
  float mx = 0.0f;                       // relu output >= 0, so 0 is identity
#pragma unroll 4
  for (int n = 0; n < 32; ++n) {
    float y = 0.f;
#pragma unroll
    for (int c = 0; c < 8; ++c) y += xs[q][c][n] * wl[c];
    mx = fmaxf(mx, fmaxf(y * scale + shift, 0.0f));
  }
  out[(size_t)(ppb + q) * 64 + o] = mx;
}

// ----------------------------------------------------------------- scatter
// canvas NHWC bf16: (B, 282, 282, 128); sweep -> ch 0..63, map -> ch 64..127
__global__ __launch_bounds__(128)
void scatter_kernel(const float* __restrict__ sweep, const float* __restrict__ mappts,
                    const float* __restrict__ feat_s, const float* __restrict__ feat_m,
                    bf16_t* __restrict__ canvas) {
  const int pp = blockIdx.x;
  const int c = threadIdx.x & 63;
  const int cloud = threadIdx.x >> 6;
  const int b = pp / P_, p = pp - b * P_;
  const float* pts = cloud ? mappts : sweep;
  const long PN = (long)P_ * N_;
  float px = pts[(size_t)b * 8 * PN + 0 * PN + (size_t)p * 32];
  float py = pts[(size_t)b * 8 * PN + 1 * PN + (size_t)p * 32];
  int xg = (int)floorf((px + RANGE_) / PILLAR_);
  int yg = (int)floorf((py + RANGE_) / PILLAR_);
  if (xg < 0 || xg >= HH || yg < 0 || yg >= HH) return;
  const float* f = (cloud ? feat_m : feat_s) + (size_t)pp * 64 + c;
  canvas[(((size_t)b * HH + xg) * HH + yg) * 128 + cloud * 64 + c] = (bf16_t)(*f);
}

// ----------------------------------------------------------- weight prep
// B-fragment lane layout: lane L -> column n = L&15, halves h -> k = (L>>4)*16+h
__global__ void prep_w1(const float* __restrict__ w, bf16_t* __restrict__ wp) {
  int idx = blockIdx.x * blockDim.x + threadIdx.x;
  if (idx >= 25 * 4 * 4 * 32 * 16) return;
  int h = idx & 15, L = (idx >> 4) & 31, j = (idx >> 9) & 3, ks = (idx >> 11) & 3;
  int tap = idx >> 13;
  int n = j * 16 + (L & 15);
  int k = ks * 32 + (L >> 4) * 16 + h;
  int ky = tap / 5, kx = tap % 5;
  wp[idx] = (bf16_t)w[((n * 128 + k) * 5 + ky) * 5 + kx];
}
__global__ void prep_w2(const float* __restrict__ w, bf16_t* __restrict__ wp) {
  int idx = blockIdx.x * blockDim.x + threadIdx.x;
  if (idx >= 16 * 2 * 32 * 16) return;
  int h = idx & 15, L = (idx >> 4) & 31, ks = (idx >> 9) & 1;
  int tap = idx >> 10;
  int n = L & 15;
  int k = ks * 32 + (L >> 4) * 16 + h;
  int ky = tap / 4, kx = tap % 4;
  wp[idx] = (bf16_t)w[((n * 64 + k) * 4 + ky) * 4 + kx];
}

// ----------------------------------------------- implicit-GEMM conv (WMMA)
// One wave computes a 16(M=y) x COUT tile for one (b, x) row; per-tap K=CIN.
template <int CIN, int COUT, int KSZ, int HIN, int OXY>
__global__ __launch_bounds__(128)
void conv_wmma(const bf16_t* __restrict__ in, const bf16_t* __restrict__ wprep,
               float* __restrict__ out) {
  constexpr int NT = COUT / 16;
  constexpr int KS = CIN / 32;
  constexpr int YT = (OXY + 15) / 16;
  const int lane = threadIdx.x & 31;
  const int wid  = (blockIdx.x * blockDim.x + threadIdx.x) >> 5;
  const int total = BSZ * OXY * YT;
  if (wid >= total) return;                      // wave-uniform: EXEC stays full
  const int yt = wid % YT;
  const int x  = (wid / YT) % OXY;
  const int b  = wid / (YT * OXY);
  const int y0 = yt * 16;
  const int half = lane >> 4;                    // K-slice select (A layout)
  const int row  = lane & 15;                    // M row carried by this lane
  int ya = y0 + row; if (ya > OXY - 1) ya = OXY - 1;   // clamp loads for tail tile

  v8f acc[NT] = {};

  for (int ky = 0; ky < KSZ; ++ky)
    for (int kx = 0; kx < KSZ; ++kx) {
      const bf16_t* arow = in + (((size_t)b * HIN + (x + ky)) * HIN + (ya + kx)) * CIN;
      const bf16_t* wrow = wprep + (size_t)(ky * KSZ + kx) * (KS * NT * 32 * 16);
#pragma unroll
      for (int ks = 0; ks < KS; ++ks) {
        const int kb = ks * 32 + half * 16;
        union { v16bf v; v8bf h[2]; } a;        // halves 0-7: k=kb.., 8-15: k=kb+8..
        a.h[0] = *(const v8bf*)(arow + kb);
        a.h[1] = *(const v8bf*)(arow + kb + 8);
#pragma unroll
        for (int j = 0; j < NT; ++j) {
          v16bf bw = *(const v16bf*)(wrow + ((size_t)(ks * NT + j) * 32 + lane) * 16);
          acc[j] = __builtin_amdgcn_wmma_f32_16x16x32_bf16(
              false, a.v, false, bw, (short)0, acc[j], false, false);
        }
      }
    }

  const int n0 = lane & 15;                      // C/D layout: lane -> column
#pragma unroll
  for (int j = 0; j < NT; ++j)
#pragma unroll
    for (int v = 0; v < 8; ++v) {
      int y = y0 + v + half * 8;                 // VGPR v: M=v (lanes 0-15) / v+8
      if (y < OXY)
        out[(((size_t)b * OXY + x) * OXY + y) * COUT + (j * 16 + n0)] = acc[j][v];
    }
}

// --------------------------------------------------------------- BN stats
__global__ __launch_bounds__(256)
void bn_stats(const float* __restrict__ x, long n, int C,
              float* __restrict__ sum, float* __restrict__ sumsq) {
  __shared__ float ssum[64], ssq[64];
  if (threadIdx.x < C) { ssum[threadIdx.x] = 0.f; ssq[threadIdx.x] = 0.f; }
  __syncthreads();
  long tid = (long)blockIdx.x * blockDim.x + threadIdx.x;
  long stride = (long)gridDim.x * blockDim.x;   // multiple of C (C in {16,64})
  int c = (int)(tid % C);                       // fixed channel per thread
  float s = 0.f, s2 = 0.f;
  for (long i = tid; i < n; i += stride) { float v = x[i]; s += v; s2 += v * v; }
  atomicAdd(&ssum[c], s); atomicAdd(&ssq[c], s2);
  __syncthreads();
  if (threadIdx.x < C) {
    atomicAdd(&sum[threadIdx.x], ssum[threadIdx.x]);
    atomicAdd(&sumsq[threadIdx.x], ssq[threadIdx.x]);
  }
}

// ------------------------------------------------- fused BN + ReLU + 2x2 pool
template <bool OUTBF>
__global__ __launch_bounds__(256)
void bn_relu_pool(const float* __restrict__ in, void* __restrict__ outv,
                  const float* __restrict__ sum, const float* __restrict__ sumsq,
                  const float* __restrict__ g, const float* __restrict__ be,
                  float invcnt, int C, int XIN, int XOUT) {
  long idx = (long)blockIdx.x * blockDim.x + threadIdx.x;
  long total = (long)BSZ * XOUT * XOUT * C;
  if (idx >= total) return;
  int c = (int)(idx % C); long t = idx / C;
  int oy = (int)(t % XOUT); t /= XOUT;
  int ox = (int)(t % XOUT); int b = (int)(t / XOUT);
  float mean = sum[c] * invcnt;
  float var  = sumsq[c] * invcnt - mean * mean;
  float scale = g[c] * rsqrtf(var + EPS_);
  float shift = be[c] - mean * scale;
  float m = 0.0f;
#pragma unroll
  for (int i = 0; i < 2; ++i)
#pragma unroll
    for (int jj = 0; jj < 2; ++jj) {
      float v = in[(((size_t)b * XIN + (2 * ox + i)) * XIN + (2 * oy + jj)) * C + c];
      m = fmaxf(m, fmaxf(v * scale + shift, 0.0f));
    }
  if constexpr (OUTBF) ((bf16_t*)outv)[idx] = (bf16_t)m;
  else                 ((float*)outv)[idx] = m;
}

// ------------------------------------------------------------------- FC1
// x is pooled2 NHWC (B,68,68,16); weight flat index follows NCHW c*4624+h*68+w
__global__ __launch_bounds__(256)
void fc1_kernel(const float* __restrict__ x, const float* __restrict__ w,
                const float* __restrict__ bias, float* __restrict__ h) {
  __shared__ float red[256];
  const int b = blockIdx.x >> 5;
  const int o = blockIdx.x & 31;
  float s = 0.f;
  for (int k = threadIdx.x; k < FC_K; k += 256) {
    int c = k / 4624; int r = k - c * 4624;
    int hh = r / 68;  int ww = r - hh * 68;
    s += w[(size_t)o * FC_K + k] * x[(((size_t)b * 68 + hh) * 68 + ww) * 16 + c];
  }
  red[threadIdx.x] = s;
  __syncthreads();
  for (int st = 128; st > 0; st >>= 1) {
    if (threadIdx.x < st) red[threadIdx.x] += red[threadIdx.x + st];
    __syncthreads();
  }
  if (threadIdx.x == 0) h[b * 32 + o] = red[0] + bias[o];  // bias cancels in BN; kept for fidelity
}

// ----------------------------------------------------- BN(batch=4)+tanh+out
__global__ void head_kernel(const float* __restrict__ h, const float* __restrict__ g,
                            const float* __restrict__ be, const float* __restrict__ wo,
                            const float* __restrict__ bo, float* __restrict__ out) {
  __shared__ float hn[4][32];
  const int o = threadIdx.x;
  if (o < 32) {
    float v0 = h[o], v1 = h[32 + o], v2 = h[64 + o], v3 = h[96 + o];
    float m = 0.25f * (v0 + v1 + v2 + v3);
    float va = 0.25f * ((v0 - m) * (v0 - m) + (v1 - m) * (v1 - m) +
                        (v2 - m) * (v2 - m) + (v3 - m) * (v3 - m));
    float sc = rsqrtf(va + EPS_) * g[o];
    hn[0][o] = tanhf((v0 - m) * sc + be[o]);
    hn[1][o] = tanhf((v1 - m) * sc + be[o]);
    hn[2][o] = tanhf((v2 - m) * sc + be[o]);
    hn[3][o] = tanhf((v3 - m) * sc + be[o]);
  }
  __syncthreads();
  if (o < 12) {
    int b = o / 3, j = o % 3;
    float s = bo[j];
#pragma unroll
    for (int k = 0; k < 32; ++k) s += hn[b][k] * wo[j * 32 + k];
    out[b * 3 + j] = s;
  }
}

// =========================================================== host launcher
extern "C" void kernel_launch(void* const* d_in, const int* in_sizes, int n_in,
                              void* d_out, int out_size, void* d_ws, size_t ws_size,
                              hipStream_t stream) {
  (void)in_sizes; (void)n_in; (void)out_size; (void)ws_size;
  const float* sweep   = (const float*)d_in[0];
  const float* mappts  = (const float*)d_in[1];
  const float* w_pfn_s = (const float*)d_in[2];
  const float* g_pfn_s = (const float*)d_in[4];
  const float* be_pfn_s= (const float*)d_in[5];
  const float* w_pfn_m = (const float*)d_in[6];
  const float* g_pfn_m = (const float*)d_in[8];
  const float* be_pfn_m= (const float*)d_in[9];
  const float* w_c1    = (const float*)d_in[10];
  const float* g_c1    = (const float*)d_in[12];
  const float* be_c1   = (const float*)d_in[13];
  const float* w_c2    = (const float*)d_in[14];
  const float* g_c2    = (const float*)d_in[16];
  const float* be_c2   = (const float*)d_in[17];
  const float* w_fc1   = (const float*)d_in[18];
  const float* b_fc1   = (const float*)d_in[19];
  const float* g_fc1   = (const float*)d_in[20];
  const float* be_fc1  = (const float*)d_in[21];
  const float* w_out   = (const float*)d_in[22];
  const float* b_out   = (const float*)d_in[23];
  float* outp = (float*)d_out;

  // -------- deterministic workspace layout (~202 MB) --------
  char* ws = (char*)d_ws;
  size_t off = 0;
  auto alloc = [&](size_t bytes) { size_t o = off; off = (off + bytes + 255) & ~(size_t)255; return o; };
  float*  stats    = (float*)(ws + alloc(512 * 4));                 // BN accumulators
  float*  hbuf     = (float*)(ws + alloc(128 * 4));                 // fc1 output
  float*  pfn_s    = (float*)(ws + alloc((size_t)BSZ * P_ * 64 * 4));
  float*  pfn_m    = (float*)(ws + alloc((size_t)BSZ * P_ * 64 * 4));
  bf16_t* canvas   = (bf16_t*)(ws + alloc((size_t)BSZ * HH * HH * 128 * 2));
  bf16_t* wprep1   = (bf16_t*)(ws + alloc((size_t)25 * 4 * 4 * 32 * 16 * 2));
  bf16_t* wprep2   = (bf16_t*)(ws + alloc((size_t)16 * 2 * 32 * 16 * 2));
  float*  conv1out = (float*)(ws + alloc((size_t)BSZ * OX1 * OX1 * 64 * 4));
  bf16_t* pooled1  = (bf16_t*)(ws + alloc((size_t)BSZ * PO1 * PO1 * 64 * 2));
  float*  conv2out = (float*)(ws + alloc((size_t)BSZ * OX2 * OX2 * 16 * 4));
  float*  pooled2  = (float*)(ws + alloc((size_t)BSZ * PO2 * PO2 * 16 * 4));

  float* s_pfn_s = stats +   0; float* q_pfn_s = stats +  64;
  float* s_pfn_m = stats + 128; float* q_pfn_m = stats + 192;
  float* s_c1    = stats + 256; float* q_c1    = stats + 320;
  float* s_c2    = stats + 384; float* q_c2    = stats + 400;

  // -------- pipeline --------
  zero_f32<<<2, 256, 0, stream>>>(stats, 512);

  dim3 pg(32, 64);
  pfn_stats<<<pg, 256, 0, stream>>>(sweep,  w_pfn_s, s_pfn_s, q_pfn_s);
  pfn_stats<<<pg, 256, 0, stream>>>(mappts, w_pfn_m, s_pfn_m, q_pfn_m);
  pfn_apply<<<BSZ * P_ / 4, 256, 0, stream>>>(sweep,  w_pfn_s, g_pfn_s, be_pfn_s,
                                              s_pfn_s, q_pfn_s, pfn_s);
  pfn_apply<<<BSZ * P_ / 4, 256, 0, stream>>>(mappts, w_pfn_m, g_pfn_m, be_pfn_m,
                                              s_pfn_m, q_pfn_m, pfn_m);

  const long canvas16 = (long)BSZ * HH * HH * 128 * 2 / 16;
  zero_u128<<<(int)((canvas16 + 255) / 256), 256, 0, stream>>>((uint4*)canvas, canvas16);
  scatter_kernel<<<BSZ * P_, 128, 0, stream>>>(sweep, mappts, pfn_s, pfn_m, canvas);

  prep_w1<<<(25 * 4 * 4 * 32 * 16) / 256, 256, 0, stream>>>(w_c1, wprep1);
  prep_w2<<<(16 * 2 * 32 * 16) / 256, 256, 0, stream>>>(w_c2, wprep2);

  {  // conv1: 128->64, 5x5, M-tiles of 16 along y
    constexpr int YT = (OX1 + 15) / 16;
    int waves = BSZ * OX1 * YT;
    conv_wmma<128, 64, 5, HH, OX1><<<(waves * 32 + 127) / 128, 128, 0, stream>>>(
        canvas, wprep1, conv1out);
  }
  bn_stats<<<1024, 256, 0, stream>>>(conv1out, C1_CNT * 64, 64, s_c1, q_c1);
  {
    long total = (long)BSZ * PO1 * PO1 * 64;
    bn_relu_pool<true><<<(int)((total + 255) / 256), 256, 0, stream>>>(
        conv1out, (void*)pooled1, s_c1, q_c1, g_c1, be_c1,
        1.0f / (float)C1_CNT, 64, OX1, PO1);
  }

  {  // conv2: 64->16, 4x4
    constexpr int YT = (OX2 + 15) / 16;
    int waves = BSZ * OX2 * YT;
    conv_wmma<64, 16, 4, PO1, OX2><<<(waves * 32 + 127) / 128, 128, 0, stream>>>(
        pooled1, wprep2, conv2out);
  }
  bn_stats<<<256, 256, 0, stream>>>(conv2out, C2_CNT * 16, 16, s_c2, q_c2);
  {
    long total = (long)BSZ * PO2 * PO2 * 16;
    bn_relu_pool<false><<<(int)((total + 255) / 256), 256, 0, stream>>>(
        conv2out, (void*)pooled2, s_c2, q_c2, g_c2, be_c2,
        1.0f / (float)C2_CNT, 16, OX2, PO2);
  }

  fc1_kernel<<<128, 256, 0, stream>>>(pooled2, w_fc1, b_fc1, hbuf);
  head_kernel<<<1, 32, 0, stream>>>(hbuf, g_fc1, be_fc1, w_out, b_out, outp);
}
